// GhostGPT_54855322304541
// MI455X (gfx1250) — matrix operand order repaired and checked by reading, not compiled, not run on hardware
//
#include <hip/hip_runtime.h>
#include <hip/hip_bf16.h>
#include <math.h>

// ---------------- model constants ----------------
#define DIM   768
#define NL    13
#define NH    12
#define VOCAB 50257
#define SEQ   512
#define BATCH 4
#define HID   3072
#define HD    64
#define NTOK  (BATCH * SEQ)   // 2048

typedef __bf16 bf16_t;
typedef bf16_t v16bf __attribute__((ext_vector_type(16)));
typedef float  v8f   __attribute__((ext_vector_type(8)));

// GEMM tiling: 128 threads = 4 waves, each wave computes 32(M) x 64(N)
#define TM   64    // block tile M
#define TN   128   // block tile N
#define TK   64    // k-slab (2 WMMA k-steps)
#define LDK  72    // padded k stride in LDS (bf16), 144B rows -> 16B-aligned chunks

// ---------------- helpers ----------------
__device__ __forceinline__ float gelu_tanh(float x) {
    const float c = 0.7978845608028654f;
    float t = tanhf(c * (x + 0.044715f * x * x * x));
    return 0.5f * x * (1.0f + t);
}

// A-fragment (16x32 bf16, LDS row-major [m][k]): lane m = L%16,
// element e -> k = (L>>4)*16 + e
__device__ __forceinline__ v16bf load_afrag(const bf16_t* base, int ldk) {
    int lane = threadIdx.x & 31;
    int m = lane & 15, half = lane >> 4;
    const bf16_t* p = base + m * ldk + half * 16;
    v16bf a;
#pragma unroll
    for (int g = 0; g < 2; ++g)
#pragma unroll
        for (int i = 0; i < 8; ++i)
            a[g * 8 + i] = p[g * 8 + i];
    return a;
}

// B-fragment (32x16, LDS n-major [n][k]): lane n = L%16,
// element e -> k = (L>>4)*16 + e
__device__ __forceinline__ v16bf load_bfrag(const bf16_t* base, int ldk) {
    int lane = threadIdx.x & 31;
    int n = lane & 15, khalf = lane >> 4;
    const bf16_t* p = base + n * ldk + khalf * 16;
    v16bf b;
#pragma unroll
    for (int i = 0; i < 16; ++i) b[i] = p[i];
    return b;
}

__device__ __forceinline__ v8f wmma_bf16(v16bf a, v16bf b, v8f c) {
    return __builtin_amdgcn_wmma_f32_16x16x32_bf16(
        false, a, false, b, (short)0, c, false, false);
}

__device__ __forceinline__ v8f v8f_zero() {
    v8f z = {0.f, 0.f, 0.f, 0.f, 0.f, 0.f, 0.f, 0.f};
    return z;
}

// Output modes for GEMM epilogues
//  0: fp32 plain                       (head)
//  1: fp32 + residual add              (out-proj with bias / mlp2)
//  2: bf16 +bias, Q-prescale n<DIM     (qkv; folds 1/sqrt(HD) into Q)
//  3: bf16 + tanh-GELU                 (mlp1)

// ---------------- embedding ----------------
__global__ void ghost_embed_kernel(const int* __restrict__ idx,
                                   const float* __restrict__ tok,
                                   const float* __restrict__ pos,
                                   float* __restrict__ x) {
    int row = blockIdx.x;               // 0..NTOK-1
    int t = row & (SEQ - 1);
    int token = idx[row];
    const float* te = tok + (size_t)token * DIM;
    const float* pe = pos + (size_t)t * DIM;
    float* xr = x + (size_t)row * DIM;
    for (int d = threadIdx.x; d < DIM; d += blockDim.x)
        xr[d] = te[d] + pe[d];
}

// ---------------- layernorm (fp32 in -> bf16 out) ----------------
__global__ void ghost_layernorm_kernel(const float* __restrict__ x,
                                       const float* __restrict__ g,
                                       const float* __restrict__ b,
                                       bf16_t* __restrict__ o) {
    __shared__ float red[64];
    int row = blockIdx.x;
    const float* xr = x + (size_t)row * DIM;
    float s = 0.f, s2 = 0.f;
    for (int d = threadIdx.x; d < DIM; d += blockDim.x) {
        float v = xr[d];
        s += v; s2 += v * v;
    }
#pragma unroll
    for (int off = 16; off >= 1; off >>= 1) {
        s  += __shfl_xor(s,  off, 32);
        s2 += __shfl_xor(s2, off, 32);
    }
    int wid = threadIdx.x >> 5, lid = threadIdx.x & 31;
    int nw = blockDim.x >> 5;
    if (lid == 0) { red[wid] = s; red[wid + 32] = s2; }
    __syncthreads();
    if (threadIdx.x == 0) {
        float a = 0.f, c = 0.f;
        for (int i = 0; i < nw; ++i) { a += red[i]; c += red[i + 32]; }
        red[0] = a; red[32] = c;
    }
    __syncthreads();
    float mean = red[0] / (float)DIM;
    float var = red[32] / (float)DIM - mean * mean;
    float rstd = rsqrtf(var + 1e-5f);
    bf16_t* orow = o + (size_t)row * DIM;
    for (int d = threadIdx.x; d < DIM; d += blockDim.x)
        orow[d] = (bf16_t)((xr[d] - mean) * rstd * g[d] + b[d]);
}

// ---------------- dense GEMM: C = A(bf16)[M,K] @ W(f32)[N,K]^T + bias ------
template <int OMODE>
__global__ __launch_bounds__(128)
void ghost_gemm_dense_kernel(const bf16_t* __restrict__ A,
                             const float* __restrict__ W,
                             const float* __restrict__ bias,
                             const float* __restrict__ res,
                             void* __restrict__ Cout,
                             int M, int N, int K) {
    __shared__ bf16_t As[2][TM * LDK];
    __shared__ bf16_t Ws[2][TN * LDK];
    const int n0 = blockIdx.x * TN;
    const int m0 = blockIdx.y * TM;
    const int tid = threadIdx.x;
    const int w = tid >> 5, wm = w >> 1, wn = w & 1;

    v8f acc[2][4];
#pragma unroll
    for (int i = 0; i < 2; ++i)
#pragma unroll
        for (int j = 0; j < 4; ++j) acc[i][j] = v8f_zero();

    auto stageA = [&](int buf, int k0) {
#pragma unroll
        for (int p = 0; p < 4; ++p) {              // 64 rows x 8 chunks = 512
            int i = tid + p * 128;
            int row = i >> 3, c8 = (i & 7) << 3;
            *(uint4*)(&As[buf][row * LDK + c8]) =
                *(const uint4*)(A + (size_t)(m0 + row) * K + k0 + c8);
        }
    };
    auto stageW = [&](int buf, int k0) {
        for (int p = 0; p < 8; ++p) {              // 128 rows x 8 chunks = 1024
            int i = tid + p * 128;
            int row = i >> 3, c8 = (i & 7) << 3;
            int n = n0 + row;
            bf16_t* d = &Ws[buf][row * LDK + c8];
            if (n < N) {
                float4 v0 = *(const float4*)(W + (size_t)n * K + k0 + c8);
                float4 v1 = *(const float4*)(W + (size_t)n * K + k0 + c8 + 4);
                d[0] = (bf16_t)v0.x; d[1] = (bf16_t)v0.y;
                d[2] = (bf16_t)v0.z; d[3] = (bf16_t)v0.w;
                d[4] = (bf16_t)v1.x; d[5] = (bf16_t)v1.y;
                d[6] = (bf16_t)v1.z; d[7] = (bf16_t)v1.w;
            } else {
#pragma unroll
                for (int e = 0; e < 8; ++e) d[e] = (bf16_t)0.f;
            }
        }
    };

    const int nk = K >> 6;          // K / 64
    stageA(0, 0); stageW(0, 0);
    __syncthreads();
    for (int t = 0; t < nk; ++t) {
        int buf = t & 1;
        if (t + 1 < nk) {
            stageA(buf ^ 1, (t + 1) << 6);
            stageW(buf ^ 1, (t + 1) << 6);
        }
        if (t + 2 < nk) {   // L2 prefetch two slabs ahead (global_prefetch_b8)
            __builtin_prefetch(A + (size_t)(m0 + (tid & 63)) * K + ((t + 2) << 6), 0, 1);
            __builtin_prefetch(W + (size_t)(n0 + tid) * K + ((t + 2) << 6), 0, 1);
        }
        const bf16_t* as = As[buf];
        const bf16_t* ws = Ws[buf];
#pragma unroll
        for (int ks = 0; ks < 2; ++ks) {
            v16bf a0 = load_afrag(as + (wm * 32 + 0)  * LDK + ks * 32, LDK);
            v16bf a1 = load_afrag(as + (wm * 32 + 16) * LDK + ks * 32, LDK);
#pragma unroll
            for (int nt = 0; nt < 4; ++nt) {
                v16bf b = load_bfrag(ws + (wn * 64 + nt * 16) * LDK + ks * 32, LDK);
                acc[0][nt] = wmma_bf16(a0, b, acc[0][nt]);
                acc[1][nt] = wmma_bf16(a1, b, acc[1][nt]);
            }
        }
        __syncthreads();
    }

    int lane = tid & 31, cn = lane & 15, half = lane >> 4;
#pragma unroll
    for (int mt = 0; mt < 2; ++mt)
#pragma unroll
        for (int nt = 0; nt < 4; ++nt) {
            int n = n0 + wn * 64 + nt * 16 + cn;
            if (n >= N) continue;
            float bv = bias[n];
#pragma unroll
            for (int r = 0; r < 8; ++r) {
                int m = m0 + wm * 32 + mt * 16 + half * 8 + r;
                float v = acc[mt][nt][r] + bv;
                if (OMODE == 1) {
                    v += res[(size_t)m * N + n];
                    ((float*)Cout)[(size_t)m * N + n] = v;
                } else if (OMODE == 0) {
                    ((float*)Cout)[(size_t)m * N + n] = v;
                } else {
                    if (OMODE == 2 && n < DIM) v *= 0.125f;  // pre-scale Q by 1/sqrt(HD)
                    if (OMODE == 3) v = gelu_tanh(v);
                    ((bf16_t*)Cout)[(size_t)m * N + n] = (bf16_t)v;
                }
            }
        }
}

// ---------------- ghost GEMM: W[n,k] = lut[gidx[n,k]] * scale[n] ----------
template <int OMODE>
__global__ __launch_bounds__(128)
void ghost_gemm_lut_kernel(const bf16_t* __restrict__ A,
                           const int* __restrict__ gidx,
                           const float* __restrict__ scale,
                           const float* __restrict__ lutg,
                           const float* __restrict__ res,
                           void* __restrict__ Cout,
                           int M, int N, int K) {
    __shared__ bf16_t As[2][TM * LDK];
    __shared__ bf16_t Ws[2][TN * LDK];
    __shared__ float lut[256];
    const int tid = threadIdx.x;
    for (int i = tid; i < 256; i += 128) lut[i] = lutg[i];

    const int n0 = blockIdx.x * TN;
    const int m0 = blockIdx.y * TM;
    const int w = tid >> 5, wm = w >> 1, wn = w & 1;

    v8f acc[2][4];
#pragma unroll
    for (int i = 0; i < 2; ++i)
#pragma unroll
        for (int j = 0; j < 4; ++j) acc[i][j] = v8f_zero();

    auto stageA = [&](int buf, int k0) {
#pragma unroll
        for (int p = 0; p < 4; ++p) {
            int i = tid + p * 128;
            int row = i >> 3, c8 = (i & 7) << 3;
            *(uint4*)(&As[buf][row * LDK + c8]) =
                *(const uint4*)(A + (size_t)(m0 + row) * K + k0 + c8);
        }
    };
    auto stageW = [&](int buf, int k0) {
        for (int p = 0; p < 8; ++p) {
            int i = tid + p * 128;
            int row = i >> 3, c8 = (i & 7) << 3;
            int n = n0 + row;
            bf16_t* d = &Ws[buf][row * LDK + c8];
            if (n < N) {
                const int* gp = gidx + (size_t)n * K + k0 + c8;
                int4 a = *(const int4*)(gp);
                int4 b = *(const int4*)(gp + 4);
                float sc = scale[n];
                d[0] = (bf16_t)(lut[a.x] * sc); d[1] = (bf16_t)(lut[a.y] * sc);
                d[2] = (bf16_t)(lut[a.z] * sc); d[3] = (bf16_t)(lut[a.w] * sc);
                d[4] = (bf16_t)(lut[b.x] * sc); d[5] = (bf16_t)(lut[b.y] * sc);
                d[6] = (bf16_t)(lut[b.z] * sc); d[7] = (bf16_t)(lut[b.w] * sc);
            } else {
#pragma unroll
                for (int e = 0; e < 8; ++e) d[e] = (bf16_t)0.f;
            }
        }
    };

    const int nk = K >> 6;
    __syncthreads();                 // lut visible
    stageA(0, 0); stageW(0, 0);
    __syncthreads();
    for (int t = 0; t < nk; ++t) {
        int buf = t & 1;
        if (t + 1 < nk) {
            stageA(buf ^ 1, (t + 1) << 6);
            stageW(buf ^ 1, (t + 1) << 6);
        }
        if (t + 2 < nk) {   // prefetch the dominant index stream into L2
            __builtin_prefetch(gidx + (size_t)(n0 + tid) * K + ((t + 2) << 6), 0, 1);
            __builtin_prefetch(A + (size_t)(m0 + (tid & 63)) * K + ((t + 2) << 6), 0, 1);
        }
        const bf16_t* as = As[buf];
        const bf16_t* ws = Ws[buf];
#pragma unroll
        for (int ks = 0; ks < 2; ++ks) {
            v16bf a0 = load_afrag(as + (wm * 32 + 0)  * LDK + ks * 32, LDK);
            v16bf a1 = load_afrag(as + (wm * 32 + 16) * LDK + ks * 32, LDK);
#pragma unroll
            for (int nt = 0; nt < 4; ++nt) {
                v16bf b = load_bfrag(ws + (wn * 64 + nt * 16) * LDK + ks * 32, LDK);
                acc[0][nt] = wmma_bf16(a0, b, acc[0][nt]);
                acc[1][nt] = wmma_bf16(a1, b, acc[1][nt]);
            }
        }
        __syncthreads();
    }

    int lane = tid & 31, cn = lane & 15, half = lane >> 4;
#pragma unroll
    for (int mt = 0; mt < 2; ++mt)
#pragma unroll
        for (int nt = 0; nt < 4; ++nt) {
            int n = n0 + wn * 64 + nt * 16 + cn;
            if (n >= N) continue;
#pragma unroll
            for (int r = 0; r < 8; ++r) {
                int m = m0 + wm * 32 + mt * 16 + half * 8 + r;
                float v = acc[mt][nt][r];
                if (OMODE == 1) {
                    v += res[(size_t)m * N + n];
                    ((float*)Cout)[(size_t)m * N + n] = v;
                } else if (OMODE == 0) {
                    ((float*)Cout)[(size_t)m * N + n] = v;
                } else {
                    if (OMODE == 3) v = gelu_tanh(v);
                    ((bf16_t*)Cout)[(size_t)m * N + n] = (bf16_t)v;
                }
            }
        }
}

// ---------------- flash attention: 1 wave per (b, h, 16-query tile) -------
// Q already pre-scaled by 1/sqrt(HD) in the QKV GEMM epilogue.
__global__ __launch_bounds__(32)
void ghost_attn_kernel(const bf16_t* __restrict__ qkv, bf16_t* __restrict__ attno) {
    __shared__ bf16_t Qs[16 * 72];
    __shared__ bf16_t Ks[32 * 72];
    __shared__ bf16_t VTs[64 * 40];   // V transposed: [hd][key]
    __shared__ bf16_t Ps[16 * 40];

    int bid = blockIdx.x;
    int qt = bid & 31;                    // SEQ/16 = 32 q-tiles
    int tmp = bid >> 5;
    int h = tmp % NH;
    int b = tmp / NH;
    int q0 = qt * 16;
    int lane = threadIdx.x;
    int half = lane >> 4, cn = lane & 15;

    const bf16_t* base = qkv + (size_t)b * SEQ * (3 * DIM);
    const bf16_t* qbase = base + h * HD;
    const bf16_t* kbase = base + DIM + h * HD;
    const bf16_t* vbase = base + 2 * DIM + h * HD;

    // stage Q tile (16 x 64) -- straight b128 copies
#pragma unroll
    for (int p = 0; p < 4; ++p) {
        int i = lane + p * 32;            // 16 rows x 8 chunks
        int row = i >> 3, c8 = (i & 7) << 3;
        *(uint4*)(&Qs[row * 72 + c8]) =
            *(const uint4*)(qbase + (size_t)(q0 + row) * (3 * DIM) + c8);
    }
    __syncthreads();
    v16bf qf[2];
    qf[0] = load_afrag(Qs + 0, 72);
    qf[1] = load_afrag(Qs + 32, 72);

    v8f o[4];
    o[0] = v8f_zero(); o[1] = v8f_zero(); o[2] = v8f_zero(); o[3] = v8f_zero();
    float rmax[8], rsum[8];
#pragma unroll
    for (int r = 0; r < 8; ++r) { rmax[r] = -3.0e38f; rsum[r] = 0.f; }

    int jend = (q0 + 15) >> 5;   // inclusive, causal
    for (int j = 0; j <= jend; ++j) {
        int k0 = j * 32;
        // stage K tile (32 keys x 64): b128 copies
#pragma unroll
        for (int p = 0; p < 8; ++p) {
            int i = lane + p * 32;        // 32 rows x 8 chunks
            int row = i >> 3, c8 = (i & 7) << 3;
            *(uint4*)(&Ks[row * 72 + c8]) =
                *(const uint4*)(kbase + (size_t)(k0 + row) * (3 * DIM) + c8);
        }
        // stage V transposed (64 x 32): vector load, scatter store
#pragma unroll
        for (int p = 0; p < 8; ++p) {
            int i = lane + p * 32;
            int row = i >> 3, c8 = (i & 7) << 3;
            union { uint4 u; bf16_t e[8]; } vv;
            vv.u = *(const uint4*)(vbase + (size_t)(k0 + row) * (3 * DIM) + c8);
#pragma unroll
            for (int e = 0; e < 8; ++e)
                VTs[(c8 + e) * 40 + row] = vv.e[e];
        }
        __syncthreads();

        // S = Q @ K^T  (16 x 32)
        v8f s0 = v8f_zero(), s1 = v8f_zero();
#pragma unroll
        for (int ks = 0; ks < 2; ++ks) {
            v16bf kb0 = load_bfrag(Ks + 0 * 72 + ks * 32, 72);
            v16bf kb1 = load_bfrag(Ks + 16 * 72 + ks * 32, 72);
            s0 = wmma_bf16(qf[ks], kb0, s0);
            s1 = wmma_bf16(qf[ks], kb1, s1);
        }

        // causal mask only on diagonal-straddling tiles
        if (k0 + 31 > q0) {
#pragma unroll
            for (int r = 0; r < 8; ++r) {
                int qrow = q0 + half * 8 + r;
                if (k0 + cn > qrow)      s0[r] = -1.0e30f;
                if (k0 + 16 + cn > qrow) s1[r] = -1.0e30f;
            }
        }
        float mloc[8];
#pragma unroll
        for (int r = 0; r < 8; ++r) mloc[r] = fmaxf(s0[r], s1[r]);
#pragma unroll
        for (int off = 8; off >= 1; off >>= 1)
#pragma unroll
            for (int r = 0; r < 8; ++r)
                mloc[r] = fmaxf(mloc[r], __shfl_xor(mloc[r], off, 32));

        float corr[8], rs[8];
#pragma unroll
        for (int r = 0; r < 8; ++r) {
            float mnew = fmaxf(rmax[r], mloc[r]);
            corr[r] = expf(rmax[r] - mnew);
            rmax[r] = mnew;
            float p0 = expf(s0[r] - mnew);
            float p1 = expf(s1[r] - mnew);
            s0[r] = p0; s1[r] = p1;
            rs[r] = p0 + p1;
        }
#pragma unroll
        for (int off = 8; off >= 1; off >>= 1)
#pragma unroll
            for (int r = 0; r < 8; ++r)
                rs[r] += __shfl_xor(rs[r], off, 32);
#pragma unroll
        for (int r = 0; r < 8; ++r) {
            rsum[r] = rsum[r] * corr[r] + rs[r];
            o[0][r] *= corr[r]; o[1][r] *= corr[r];
            o[2][r] *= corr[r]; o[3][r] *= corr[r];
            int m = half * 8 + r;
            Ps[m * 40 + cn]      = (bf16_t)s0[r];
            Ps[m * 40 + 16 + cn] = (bf16_t)s1[r];
        }
        __syncthreads();

        v16bf pf = load_afrag(Ps, 40);
#pragma unroll
        for (int nt = 0; nt < 4; ++nt) {
            v16bf vb = load_bfrag(VTs + (nt * 16) * 40, 40);
            o[nt] = wmma_bf16(pf, vb, o[nt]);
        }
        __syncthreads();
    }

    // epilogue: O / rowsum -> attno (bf16, [NTOK, DIM])
#pragma unroll
    for (int r = 0; r < 8; ++r) {
        int m = q0 + half * 8 + r;
        float inv = 1.0f / rsum[r];
        bf16_t* orow = attno + (size_t)(b * SEQ + m) * DIM + h * HD;
#pragma unroll
        for (int nt = 0; nt < 4; ++nt)
            orow[nt * 16 + cn] = (bf16_t)(o[nt][r] * inv);
    }
}

// ---------------- host launcher ----------------
extern "C" void kernel_launch(void* const* d_in, const int* in_sizes, int n_in,
                              void* d_out, int out_size, void* d_ws, size_t ws_size,
                              hipStream_t stream) {
    (void)in_sizes; (void)n_in; (void)out_size; (void)ws_size;

    const float* lut      = (const float*)d_in[0];
    const float* tok_emb  = (const float*)d_in[1];
    const float* pos_emb  = (const float*)d_in[2];
    const float* ln1_g    = (const float*)d_in[3];
    const float* ln1_b    = (const float*)d_in[4];
    const float* in_w     = (const float*)d_in[5];
    const float* in_b     = (const float*)d_in[6];
    const float* out_w    = (const float*)d_in[7];
    const float* out_b    = (const float*)d_in[8];
    const float* ln2_g    = (const float*)d_in[9];
    const float* ln2_b    = (const float*)d_in[10];
    const float* mlp1_s   = (const float*)d_in[11];
    const float* mlp2_s   = (const float*)d_in[12];
    const float* lnf_g    = (const float*)d_in[13];
    const float* lnf_b    = (const float*)d_in[14];
    const float* head_s   = (const float*)d_in[15];
    const int* mlp1_idx   = (const int*)d_in[16];
    const int* mlp2_idx   = (const int*)d_in[17];
    const int* head_idx   = (const int*)d_in[18];
    const int* tok_idx    = (const int*)d_in[19];

    // workspace carve-up
    float*  x     = (float*)d_ws;                                   // [NTOK, DIM] f32
    bf16_t* h     = (bf16_t*)(x + (size_t)NTOK * DIM);              // [NTOK, DIM] bf16
    bf16_t* qkv   = h + (size_t)NTOK * DIM;                         // [NTOK, 3*DIM] bf16
    bf16_t* attno = qkv + (size_t)NTOK * 3 * DIM;                   // [NTOK, DIM] bf16
    bf16_t* mlph  = attno + (size_t)NTOK * DIM;                     // [NTOK, HID] bf16
    float*  out   = (float*)d_out;                                  // [NTOK, VOCAB] f32

    ghost_embed_kernel<<<NTOK, 256, 0, stream>>>(tok_idx, tok_emb, pos_emb, x);

    for (int l = 0; l < NL; ++l) {
        ghost_layernorm_kernel<<<NTOK, 256, 0, stream>>>(x, ln1_g + l * DIM, ln1_b + l * DIM, h);
        // QKV projection -> bf16 (+bias, Q pre-scaled)
        ghost_gemm_dense_kernel<2><<<dim3(3 * DIM / TN, NTOK / TM), 128, 0, stream>>>(
            h, in_w + (size_t)l * 3 * DIM * DIM, in_b + (size_t)l * 3 * DIM,
            nullptr, qkv, NTOK, 3 * DIM, DIM);
        ghost_attn_kernel<<<BATCH * NH * (SEQ / 16), 32, 0, stream>>>(qkv, attno);
        // out projection -> fp32, +bias, +residual (x)
        ghost_gemm_dense_kernel<1><<<dim3(DIM / TN, NTOK / TM), 128, 0, stream>>>(
            attno, out_w + (size_t)l * DIM * DIM, out_b + (size_t)l * DIM,
            x, x, NTOK, DIM, DIM);
        ghost_layernorm_kernel<<<NTOK, 256, 0, stream>>>(x, ln2_g + l * DIM, ln2_b + l * DIM, h);
        // MLP1 (ghost) -> bf16 + GELU
        ghost_gemm_lut_kernel<3><<<dim3(HID / TN, NTOK / TM), 128, 0, stream>>>(
            h, mlp1_idx + (size_t)l * HID * DIM, mlp1_s + (size_t)l * HID, lut,
            nullptr, mlph, NTOK, HID, DIM);
        // MLP2 (ghost) -> fp32 + residual (x)
        ghost_gemm_lut_kernel<1><<<dim3(DIM / TN, NTOK / TM), 128, 0, stream>>>(
            mlph, mlp2_idx + (size_t)l * DIM * HID, mlp2_s + (size_t)l * DIM, lut,
            x, x, NTOK, DIM, HID);
    }

    ghost_layernorm_kernel<<<NTOK, 256, 0, stream>>>(x, lnf_g, lnf_b, h);
    // LM head (ghost) -> fp32
    ghost_gemm_lut_kernel<0><<<dim3((VOCAB + TN - 1) / TN, NTOK / TM), 128, 0, stream>>>(
        h, head_idx, head_s, lut, nullptr, out, NTOK, VOCAB, DIM);
}
